// MultiHeadSelfAttention_78254304133356
// MI455X (gfx1250) — matrix operand order repaired and checked by reading, not compiled, not run on hardware
//
#include <hip/hip_runtime.h>
#include <hip/hip_bf16.h>

// MI455X (gfx1250) multi-head self-attention:
//   fp32 -> f16 convert (x transposed), WMMA f32_16x16x32_f16 GEMMs,
//   flash-attention online softmax. Operand staging uses BOTH CDNA5 DMA
//   paths: TDM (tensor_load_to_lds + s_wait_tensorcnt) for the contiguous
//   K tile, and GLOBAL_LOAD_ASYNC_TO_LDS_B128 (+ s_wait_asynccnt) for the
//   strided V / W / X tiles. f32 accumulate, f32 output.
//
// Dims: B=2, S=2048, C=IN=OUT=KEY=512, H=8, hd=64.

typedef __attribute__((ext_vector_type(16))) _Float16 v16h;
typedef __attribute__((ext_vector_type(8)))  float    v8f;
typedef __attribute__((ext_vector_type(4)))  unsigned v4u;
typedef __attribute__((ext_vector_type(8)))  int      v8i;
typedef __attribute__((ext_vector_type(4)))  int      v4i;

#define WMMA_F32_F16(a, b, c) \
  __builtin_amdgcn_wmma_f32_16x16x32_f16(false, (a), false, (b), (short)0, (c), false, false)

union AF { unsigned u[8]; v16h v; };

// Async contiguous 16B-per-lane copy: global -> LDS, tracked on ASYNCcnt.
__device__ __forceinline__ void async_copy_b128(unsigned lds_off, const void* gaddr) {
  asm volatile("global_load_async_to_lds_b128 %0, %1, off"
               :: "v"(lds_off), "v"((unsigned long long)(size_t)gaddr)
               : "memory");
}
__device__ __forceinline__ void wait_async_all() {
  asm volatile("s_wait_asynccnt 0" ::: "memory");
}
// Truncate a generic pointer to a shared var to its LDS byte offset.
__device__ __forceinline__ unsigned lds_off_of(const void* p) {
  return (unsigned)(size_t)p;
}

// TDM: 1-D DMA of `bytes` (multiple of 4) from contiguous global to LDS.
// D# per ISA 8.3/8.4: group0 = {count=1, lds_addr, global_addr[56:0], type=2},
// group1 = {data_size=2B, tensor dims/strides, tile_dim0 = bytes/2 elements}.
// This toolchain exposes the 6-arg builtin (g0, g1, g2, g3, g4, cpol).
__device__ __forceinline__ void tdm_load_1d(unsigned lds_off, const void* gaddr,
                                            unsigned bytes) {
  const unsigned long long ga = (unsigned long long)(size_t)gaddr;
  v4u g0;
  g0[0] = 1u;                                               // count=1, user mode
  g0[1] = lds_off;                                          // lds_addr [63:32]
  g0[2] = (unsigned)(ga & 0xFFFFFFFFu);                     // global_addr [95:64]
  g0[3] = (unsigned)((ga >> 32) & 0x1FFFFFFu) | (2u << 30); // addr[56:32] | type=2
  const unsigned elems = bytes >> 1;                        // data_size = 2B
  v8i g1;
  g1[0] = (1 << 16);                // data_size code 1 (2B); no mask/pad/iterate
  g1[1] = (int)((elems & 0xFFFFu) << 16);    // tensor_dim0[15:0] -> bits[63:48]
  g1[2] = (int)((elems >> 16) | (1u << 16)); // tensor_dim0[31:16]; tensor_dim1=1
  g1[3] = (int)((elems & 0xFFFFu) << 16);    // tile_dim0 (16-bit) -> bits[127:112]
  g1[4] = 0;                        // tile_dim1/2 unused (1-D)
  g1[5] = (int)elems;               // tensor_dim0_stride low32
  g1[6] = 0;
  g1[7] = 0;
  const v4i gz4 = {0, 0, 0, 0};
  const v8i gz8 = {0, 0, 0, 0, 0, 0, 0, 0};
  __builtin_amdgcn_tensor_load_to_lds(g0, g1, gz4, gz4, gz8, 0);
}

// A fragment, 16x32 f16 (M x K), row-major storage (stride in halfs).
// ISA 7.12.2: lane L: m=L&15, hi=L>>4; VGPR j<4: K=8hi+2j; j>=4: K=16+8hi+2(j-4).
__device__ __forceinline__ v16h load_frag_a_rowmajor(const _Float16* base, int stride) {
  const int lane = threadIdx.x & 31;
  const int m = lane & 15, hi = lane >> 4;
  const _Float16* p = base + m * stride;
  AF f;
#pragma unroll
  for (int j = 0; j < 8; ++j) {
    const int koff = (j < 4) ? (8 * hi + 2 * j) : (16 + 8 * hi + 2 * (j - 4));
    f.u[j] = *(const unsigned*)(p + koff);
  }
  return f.v;
}

// B fragment, 32x16 f16 (K x N), from COLUMN-major storage Bt[n][k].
// lane L: n=L&15, hi=L>>4; half j<8: Bt[n][8*hi + j]; j>=8: Bt[n][16 + 8*hi + (j-8)].
__device__ __forceinline__ v16h load_frag_b_colmajor(const _Float16* base, int stride) {
  const int lane = threadIdx.x & 31;
  const int n = lane & 15, hi = lane >> 4;
  const _Float16* p = base + n * stride + 8 * hi;
  AF f;
#pragma unroll
  for (int j = 0; j < 8; ++j) f.u[j] = *(const unsigned*)(p + ((j < 4) ? 2 * j : 16 + 2 * (j - 4)));
  return f.v;
}

__global__ void cvt_f32_to_f16(const float* __restrict__ in, _Float16* __restrict__ out, int n) {
  const int i = blockIdx.x * blockDim.x + threadIdx.x;
  if (i < n) out[i] = (_Float16)in[i];
}

// x (B, C=512, S=2048) fp32  ->  xT (B, S, C) f16  (coalesced reads).
__global__ void cvt_transpose_x(const float* __restrict__ x, _Float16* __restrict__ xT) {
  const int i = blockIdx.x * 256 + threadIdx.x;   // 0 .. 2M-1
  const int b = i >> 20;                          // 512*2048 = 1<<20
  const int rem = i & ((1 << 20) - 1);
  const int c = rem >> 11;
  const int s = rem & 2047;
  xT[((size_t)b << 20) + (size_t)s * 512 + c] = (_Float16)x[i];
}

// C[o][s] = (sum_c W[o][c] * XT[b][s][c] + bias[o]) * scale
// W: 512x512 row-major. XT: (B, 2048, 512) s-major (transposed activations).
// 256 thr = 8 waves; block tile 128(M) x 64(N); wave 32x32; K-step 32.
// LDS staging via async b128 copies.
// LAYOUT 0: out[b][o][s].  LAYOUT 1: out[b][h][s][d] (head-transposed).
template <typename OutT, int LAYOUT>
__global__ __launch_bounds__(256) void proj_gemm_kernel(
    const _Float16* __restrict__ W, const _Float16* __restrict__ XTall,
    const float* __restrict__ bias, OutT* __restrict__ Oall, float scale) {
  constexpr int Kdim = 512, N = 2048;
  __shared__ alignas(16) _Float16 Wt[128][32];  // [m][k] row-major
  __shared__ alignas(16) _Float16 Xt[64][32];   // [n=s][k] col-major-for-B

  const int tid = threadIdx.x;
  const int b = blockIdx.z;
  const int o0 = blockIdx.x * 128;
  const int s0 = blockIdx.y * 64;
  const char* Wb = (const char*)W;
  const char* Xb = (const char*)(XTall + (size_t)b * N * Kdim);
  const unsigned wt0 = lds_off_of(&Wt[0][0]);
  const unsigned xt0 = lds_off_of(&Xt[0][0]);

  const int wave = tid >> 5;
  const int om = (wave >> 1) * 32;
  const int on = (wave & 1) * 32;

  const v8f vzero = {0.f, 0.f, 0.f, 0.f, 0.f, 0.f, 0.f, 0.f};
  v8f acc[2][2];
#pragma unroll
  for (int i = 0; i < 2; ++i)
#pragma unroll
    for (int j = 0; j < 2; ++j) acc[i][j] = vzero;

  for (int kk0 = 0; kk0 < Kdim; kk0 += 32) {
    // W tile: 128 rows x 64B = 512 b128 chunks, 2 per thread.
#pragma unroll
    for (int i = 0; i < 2; ++i) {
      const int id = tid + 256 * i;
      const int row = id >> 2, c4 = id & 3;
      async_copy_b128(wt0 + id * 16,
                      Wb + (size_t)(o0 + row) * 1024 + kk0 * 2 + c4 * 16);
    }
    // X tile: 64 rows x 64B = 256 b128 chunks, 1 per thread.
    {
      const int row = tid >> 2, c4 = tid & 3;
      async_copy_b128(xt0 + tid * 16,
                      Xb + (size_t)(s0 + row) * 1024 + kk0 * 2 + c4 * 16);
    }
    // Prefetch next W tile while async copies are in flight.
    if (kk0 + 32 < Kdim)
      __builtin_prefetch(Wb + (size_t)(o0 + (tid >> 1)) * 1024 + (kk0 + 32) * 2 +
                             (tid & 1) * 64, 0, 0);
    wait_async_all();
    __syncthreads();

    const v16h a0 = load_frag_a_rowmajor(&Wt[om][0], 32);
    const v16h a1 = load_frag_a_rowmajor(&Wt[om + 16][0], 32);
    const v16h b0 = load_frag_b_colmajor(&Xt[on][0], 32);
    const v16h b1 = load_frag_b_colmajor(&Xt[on + 16][0], 32);
    acc[0][0] = WMMA_F32_F16(a0, b0, acc[0][0]);
    acc[0][1] = WMMA_F32_F16(a0, b1, acc[0][1]);
    acc[1][0] = WMMA_F32_F16(a1, b0, acc[1][0]);
    acc[1][1] = WMMA_F32_F16(a1, b1, acc[1][1]);
    __syncthreads();
  }

  // Epilogue: C/D layout lane L holds col n=L&15, rows 8*(L>>4)+r.
  const int lane = tid & 31;
  const int n = lane & 15, hi = lane >> 4;
  const size_t bo = (size_t)b * 512 * 2048;
#pragma unroll
  for (int mt = 0; mt < 2; ++mt) {
#pragma unroll
    for (int r = 0; r < 8; ++r) {
      const int o = o0 + om + mt * 16 + 8 * hi + r;
      const float bv = bias[o];
#pragma unroll
      for (int nt = 0; nt < 2; ++nt) {
        const int s = s0 + on + nt * 16 + n;
        const float val = (acc[mt][nt][r] + bv) * scale;
        size_t idx;
        if (LAYOUT == 0)
          idx = bo + (size_t)o * 2048 + s;
        else
          idx = bo + (size_t)(o >> 6) * (2048 * 64) + (size_t)s * 64 + (o & 63);
        Oall[idx] = (OutT)val;
      }
    }
  }
}

// Flash attention per (b,h). Grid (S/64, B*H), 128 thr = 4 waves; each wave
// owns 16 query rows; key tiles of 32.
//   Q,K: (bh, s, 64) head-transposed -> K tile is a contiguous 4KB slab,
//        DMA'd by the TDM (wave 0 issues, TENSORcnt wait, barrier publishes).
//   V:   (b, o, s) natural -> VtT[d][s] staged via async b128 chunks.
// Output stored transposed (b, s, o) f16 (coalesced) = transposed-B layout
// consumed by the wo GEMM.
__global__ __launch_bounds__(128) void attn_kernel(
    const _Float16* __restrict__ qT, const _Float16* __restrict__ kT,
    const _Float16* __restrict__ vN, _Float16* __restrict__ athT) {
  constexpr int S = 2048;
  __shared__ alignas(16) _Float16 Kt[32][64];   // [s][d]
  __shared__ alignas(16) _Float16 VtT[64][32];  // [d][s]
  __shared__ _Float16 plds[4][16][34];          // wave-private P (C->A layout)

  const int tid = threadIdx.x;
  const int wave = tid >> 5;
  const int lane = tid & 31;
  const int n = lane & 15, hi = lane >> 4;

  const int bh = blockIdx.y;
  const int qt0 = blockIdx.x * 64 + wave * 16;

  const _Float16* qTb = qT + ((size_t)bh * S + qt0) * 64;
  const char* kgb = (const char*)(kT + (size_t)bh * S * 64);
  const char* vgb = (const char*)(vN + (size_t)bh * 64 * S);
  const unsigned kt0 = lds_off_of(&Kt[0][0]);
  const unsigned vt0 = lds_off_of(&VtT[0][0]);

  // Q fragments (16 rows x d=64) held in registers for the whole pass.
  const v16h qa0 = load_frag_a_rowmajor(qTb, 64);       // d 0..31
  const v16h qa1 = load_frag_a_rowmajor(qTb + 32, 64);  // d 32..63

  const v8f vzero = {0.f, 0.f, 0.f, 0.f, 0.f, 0.f, 0.f, 0.f};
  float m_row[8], l_row[8];
  v8f o_acc[4];
#pragma unroll
  for (int r = 0; r < 8; ++r) { m_row[r] = -1e30f; l_row[r] = 0.f; }
#pragma unroll
  for (int d4 = 0; d4 < 4; ++d4) o_acc[d4] = vzero;

  for (int s0 = 0; s0 < S; s0 += 32) {
    // K tile: contiguous 4KB -> single TDM descriptor from wave 0.
    if (wave == 0) {
      tdm_load_1d(kt0, kgb + (size_t)s0 * 128, 4096);
      __builtin_amdgcn_s_wait_tensorcnt(0);
    }
    // V tile: 64 rows (d) x 64B (s-pairs) = 256 b128 chunks, 2 per thread.
#pragma unroll
    for (int i = 0; i < 2; ++i) {
      const int id = tid + 128 * i;  // 0..255
      const int d = id >> 2, c4 = id & 3;
      async_copy_b128(vt0 + id * 16,
                      vgb + (size_t)d * 4096 + (size_t)s0 * 2 + c4 * 16);
    }
    wait_async_all();
    __syncthreads();

    // Scores: 16 (query rows) x 32 (keys): 2 n-subtiles, K=64 in 2 WMMA steps.
    v8f sc[2];
#pragma unroll
    for (int ns = 0; ns < 2; ++ns) {
      const v16h kb0 = load_frag_b_colmajor(&Kt[ns * 16][0], 64);   // d 0..31
      const v16h kb1 = load_frag_b_colmajor(&Kt[ns * 16][32], 64);  // d 32..63
      v8f c = vzero;
      c = WMMA_F32_F16(qa0, kb0, c);
      c = WMMA_F32_F16(qa1, kb1, c);
      sc[ns] = c;
    }

    // Online softmax. Row M=8*hi+r lives in 16 lanes sharing `hi`.
    float mx[8], alpha[8], psum[8];
#pragma unroll
    for (int r = 0; r < 8; ++r) mx[r] = fmaxf(sc[0][r], sc[1][r]);
#pragma unroll
    for (int off = 8; off >= 1; off >>= 1)
#pragma unroll
      for (int r = 0; r < 8; ++r) mx[r] = fmaxf(mx[r], __shfl_xor(mx[r], off, 32));
#pragma unroll
    for (int r = 0; r < 8; ++r) {
      const float nm = fmaxf(m_row[r], mx[r]);
      alpha[r] = __expf(m_row[r] - nm);
      m_row[r] = nm;
      const float p0 = __expf(sc[0][r] - nm);
      const float p1 = __expf(sc[1][r] - nm);
      sc[0][r] = p0;
      sc[1][r] = p1;
      psum[r] = p0 + p1;
    }
#pragma unroll
    for (int off = 8; off >= 1; off >>= 1)
#pragma unroll
      for (int r = 0; r < 8; ++r) psum[r] += __shfl_xor(psum[r], off, 32);
#pragma unroll
    for (int r = 0; r < 8; ++r) l_row[r] = l_row[r] * alpha[r] + psum[r];
#pragma unroll
    for (int d4 = 0; d4 < 4; ++d4)
#pragma unroll
      for (int r = 0; r < 8; ++r) o_acc[d4][r] *= alpha[r];

    // P: C-layout -> A-layout via wave-private LDS (in-order per wave).
#pragma unroll
    for (int r = 0; r < 8; ++r) {
      plds[wave][8 * hi + r][n]      = (_Float16)sc[0][r];
      plds[wave][8 * hi + r][16 + n] = (_Float16)sc[1][r];
    }
    const v16h pa = load_frag_a_rowmajor(&plds[wave][0][0], 34);

    // O += P (16x32) x V (32x64), 4 n-subtiles of 16.
#pragma unroll
    for (int d4 = 0; d4 < 4; ++d4) {
      const v16h vb = load_frag_b_colmajor(&VtT[d4 * 16][0], 32);
      o_acc[d4] = WMMA_F32_F16(pa, vb, o_acc[d4]);
    }
    __syncthreads();
  }

  // Normalize; store transposed (b, s, o=h*64+d) f16: coalesced u16 stores.
  const int b = bh >> 3, h = bh & 7;
  float inv[8];
#pragma unroll
  for (int r = 0; r < 8; ++r) inv[r] = 1.0f / l_row[r];
  const size_t bo = (size_t)b * 512 * 2048;
#pragma unroll
  for (int d4 = 0; d4 < 4; ++d4) {
    const int o = h * 64 + d4 * 16 + n;
#pragma unroll
    for (int r = 0; r < 8; ++r) {
      const int srow = qt0 + 8 * hi + r;
      athT[bo + (size_t)srow * 512 + o] = (_Float16)(o_acc[d4][r] * inv[r]);
    }
  }
}

extern "C" void kernel_launch(void* const* d_in, const int* in_sizes, int n_in,
                              void* d_out, int out_size, void* d_ws, size_t ws_size,
                              hipStream_t stream) {
  const float* x  = (const float*)d_in[0];
  const float* wq = (const float*)d_in[1];
  const float* bq = (const float*)d_in[2];
  const float* wk = (const float*)d_in[3];
  const float* bk = (const float*)d_in[4];
  const float* wv = (const float*)d_in[5];
  const float* bv = (const float*)d_in[6];
  const float* wo = (const float*)d_in[7];
  const float* bo = (const float*)d_in[8];
  float* out = (float*)d_out;

  const size_t NX = (size_t)2 * 512 * 2048;  // activation tensor elems
  const size_t NW = (size_t)512 * 512;       // weight elems

  // Workspace: 22 MiB of f16 scratch.
  char* w = (char*)d_ws;
  _Float16* xT  = (_Float16*)w; w += NX * 2;  // (B, S, C)
  _Float16* wqh = (_Float16*)w; w += NW * 2;
  _Float16* wkh = (_Float16*)w; w += NW * 2;
  _Float16* wvh = (_Float16*)w; w += NW * 2;
  _Float16* woh = (_Float16*)w; w += NW * 2;
  _Float16* qTh = (_Float16*)w; w += NX * 2;  // (BH, S, 64), scaled 1/8
  _Float16* kTh = (_Float16*)w; w += NX * 2;  // (BH, S, 64)
  _Float16* vNh = (_Float16*)w; w += NX * 2;  // (B, 512, S) natural
  _Float16* ath = (_Float16*)w; w += NX * 2;  // (B, S, 512) transposed

  cvt_transpose_x<<<(int)(NX / 256), 256, 0, stream>>>(x, xT);
  cvt_f32_to_f16<<<(int)(NW / 256), 256, 0, stream>>>(wq, wqh, (int)NW);
  cvt_f32_to_f16<<<(int)(NW / 256), 256, 0, stream>>>(wk, wkh, (int)NW);
  cvt_f32_to_f16<<<(int)(NW / 256), 256, 0, stream>>>(wv, wvh, (int)NW);
  cvt_f32_to_f16<<<(int)(NW / 256), 256, 0, stream>>>(wo, woh, (int)NW);

  const dim3 pgrid(4, 32, 2);  // M/128, N/64, B
  // Q/K head-transposed (b,h,s,d), Q folds in 1/sqrt(64); V natural (b,o,s).
  proj_gemm_kernel<_Float16, 1><<<pgrid, 256, 0, stream>>>(wqh, xT, bq, qTh, 0.125f);
  proj_gemm_kernel<_Float16, 1><<<pgrid, 256, 0, stream>>>(wkh, xT, bk, kTh, 1.0f);
  proj_gemm_kernel<_Float16, 0><<<pgrid, 256, 0, stream>>>(wvh, xT, bv, vNh, 1.0f);

  attn_kernel<<<dim3(32, 16), 128, 0, stream>>>(qTh, kTh, vNh, ath);

  // Final projection: consumes ath (b, s, c) as transposed-B, f32 natural out.
  proj_gemm_kernel<float, 0><<<pgrid, 256, 0, stream>>>(woh, ath, bo, out, 1.0f);
}
